// AttentionFlowLayer_64261300683116
// MI455X (gfx1250) — matrix operand order repaired and checked by reading, not compiled, not run on hardware
//
#include <hip/hip_runtime.h>

// ---------------------------------------------------------------------------
// AttentionFlowLayer for MI455X (gfx1250, wave32, WMMA)
// Shapes: N=16, P=2048, A=256, D=256.  Output G: (16, 2048, 1024) f32.
// Both 2048x256x256 GEMM chains via v_wmma_f32_16x16x32_bf16; softmax/max/
// accumulation in f32.  HBM-bound on the 134MB output -> NT stores.
// U staged into LDS via double-buffered global_load_async_to_lds_b128
// (inline asm; ASYNCcnt), converted to bf16 (row-major + transposed).
// ---------------------------------------------------------------------------

#define NN 16
#define PP 2048
#define AA 256
#define DD 256
#define PT 64          // p-rows per workgroup (4 waves x 16-row strips)
#define NWGP (PP / PT) // 32 p-tiles per n
#define TPB 128

typedef __attribute__((ext_vector_type(16))) __bf16 v16bf;
typedef __attribute__((ext_vector_type(8)))  float  v8f;

struct __align__(16) Q16 { unsigned x, y, z, w; };
union Frag { v16bf v; unsigned short u[16]; Q16 q[2]; };

// ---- async global->LDS copy (CDNA5 ASYNCcnt path) via inline asm ----------
// VDST operand carries the 32-bit LDS byte offset.
typedef __attribute__((address_space(3))) char* lds_cp;
static __device__ __forceinline__ unsigned lds_off32(void* p) {
  return (unsigned)(__UINTPTR_TYPE__)(lds_cp)(char*)p;  // addrspacecast -> u32
}
static __device__ __forceinline__ void copy16_async(void* lds, const void* g) {
  asm volatile("global_load_async_to_lds_b128 %0, %1, off"
               :
               : "v"(lds_off32(lds)),
                 "v"((unsigned long long)(__UINTPTR_TYPE__)g)
               : "memory");
}
#if __has_builtin(__builtin_amdgcn_s_wait_asynccnt)
#define WAIT_ASYNC(n) __builtin_amdgcn_s_wait_asynccnt(n)
#else
#define WAIT_ASYNC(n) asm volatile("s_wait_asynccnt %0" ::"i"(n) : "memory")
#endif

// float -> bf16 bits, round-to-nearest-even
static __device__ __forceinline__ unsigned short f2bf_bits(float f) {
  union { float f; unsigned u; } c{f};
  unsigned r = c.u + 0x7fffu + ((c.u >> 16) & 1u);
  return (unsigned short)(r >> 16);
}

// wave32 XOR-lane shuffle via ds_swizzle (group-of-32: imm = xor<<10 | and=0x1f)
template <int M>
static __device__ __forceinline__ float swz_xor(float v) {
  int x = __builtin_amdgcn_ds_swizzle(__builtin_bit_cast(int, v),
                                      (M << 10) | 0x1f);
  return __builtin_bit_cast(float, x);
}
// reductions confined to 16-lane halves (masks 1,2,4,8 never cross lane 16)
static __device__ __forceinline__ float red_max16(float v) {
  v = fmaxf(v, swz_xor<1>(v)); v = fmaxf(v, swz_xor<2>(v));
  v = fmaxf(v, swz_xor<4>(v)); v = fmaxf(v, swz_xor<8>(v));
  return v;
}
static __device__ __forceinline__ float red_sum16(float v) {
  v += swz_xor<1>(v); v += swz_xor<2>(v);
  v += swz_xor<4>(v); v += swz_xor<8>(v);
  return v;
}

static __device__ __forceinline__ v8f wmma_bf16(const Frag& a, const Frag& b,
                                                v8f c) {
  return __builtin_amdgcn_wmma_f32_16x16x32_bf16(
      /*neg_a=*/false, a.v, /*neg_b=*/false, b.v,
      /*c_mod=*/(short)0, c, /*reuse_a=*/false, /*reuse_b=*/false);
}

// LDS layout (bytes), total 301312 < 320KB WGP LDS.
// OFF_PROBS doubles as the 2x16KB f32 async staging area during U staging
// (probs is only live after GEMM 1).
#define OFF_UB     0              // ushort[256][256]  U bf16 row-major
#define OFF_UT     131072         // ushort[256][256]  U^T bf16
#define OFF_PROBS  262144         // ushort[64][256]   softmax probs bf16
#define OFF_BH     294912         // float [4][256]    per-wave b*H partials
#define OFF_UA     299008         // float [256]       U . wa
#define OFF_HC     300032         // float [64]        H . wc
#define OFF_WP     300288         // float [256]       wp
#define SMEM_BYTES 301312

#define ROWS_PER_CHUNK 16                       // 16 rows x 256 f32 = 16KB
#define NCHUNK (AA / ROWS_PER_CHUNK)            // 16
#define SEGS_PER_THREAD ((ROWS_PER_CHUNK * DD * 4 / 16) / TPB)  // 8 x 16B

__global__ __launch_bounds__(TPB, 1)
void pass1(const float* __restrict__ H, const float* __restrict__ U,
           const float* __restrict__ W, float* __restrict__ out,
           float* __restrict__ partial) {
  extern __shared__ char smem[];
  unsigned short* Ub    = (unsigned short*)(smem + OFF_UB);
  unsigned short* Ut    = (unsigned short*)(smem + OFF_UT);
  unsigned short* probs = (unsigned short*)(smem + OFF_PROBS);
  float* bh_s = (float*)(smem + OFF_BH);
  float* ua_s = (float*)(smem + OFF_UA);
  float* hc_s = (float*)(smem + OFF_HC);
  float* wp_s = (float*)(smem + OFF_WP);

  const int n   = blockIdx.y;
  const int pt  = blockIdx.x;
  const int p0  = pt * PT;
  const int tid = threadIdx.x;
  const int lane = tid & 31;
  const int wave = tid >> 5;
  const int L  = lane & 15;   // row/col index within a 16-wide fragment
  const int hi = lane >> 4;   // which K-half this lane carries

  const float* Hn = H + (size_t)n * PP * DD;
  const float* Un = U + (size_t)n * AA * DD;

  // ---- small staging: wp, ua = U.wa, hc = H.wc ----------------------------
  for (int d = tid; d < DD; d += TPB) wp_s[d] = W[512 + d];
  for (int a = tid; a < AA; a += TPB) {
    float s = 0.f;
    const float* ur = Un + (size_t)a * DD;
    for (int d = 0; d < DD; ++d) s += ur[d] * W[256 + d];
    ua_s[a] = s;
  }
  if (tid < PT) {
    float s = 0.f;
    const float* hr = Hn + (size_t)(p0 + tid) * DD;
    for (int d = 0; d < DD; ++d) s += hr[d] * W[d];
    hc_s[tid] = s;
  }

  // ---- U -> LDS bf16 (row-major Ub + transposed Ut), double-buffered ------
  // async DMA chunk c+1 while converting chunk c out of the staging buffer.
  {
    char* stage0 = smem + OFF_PROBS;                 // 16KB
    char* stage1 = smem + OFF_PROBS + 16384;         // 16KB
    auto issue = [&](int c) {
      char* dst = (c & 1) ? stage1 : stage0;
      const char* src = (const char*)(Un + (size_t)c * ROWS_PER_CHUNK * DD);
#pragma unroll
      for (int i = 0; i < SEGS_PER_THREAD; ++i) {
        int seg = tid + i * TPB;
        copy16_async(dst + seg * 16, src + seg * 16);
      }
    };
    issue(0);
    for (int c = 0; c < NCHUNK; ++c) {
      if (c + 1 < NCHUNK) {
        issue(c + 1);
        WAIT_ASYNC(SEGS_PER_THREAD);  // chunk c complete, c+1 may be in flight
      } else {
        WAIT_ASYNC(0);
      }
      __syncthreads();  // chunk c visible to all waves
      const float* stage = (const float*)((c & 1) ? stage1 : stage0);
#pragma unroll
      for (int i = 0; i < (ROWS_PER_CHUNK * DD / 2) / TPB; ++i) {  // 16 pairs
        int q = tid + i * TPB;
        int a_loc = q >> 7;              // 128 pairs per row
        int d0 = (q & 127) * 2;
        int a = c * ROWS_PER_CHUNK + a_loc;
        unsigned short b0 = f2bf_bits(stage[a_loc * DD + d0]);
        unsigned short b1 = f2bf_bits(stage[a_loc * DD + d0 + 1]);
        ((unsigned*)Ub)[(a * DD + d0) >> 1] = (unsigned)b0 | ((unsigned)b1 << 16);
        Ut[d0 * AA + a]       = b0;
        Ut[(d0 + 1) * AA + a] = b1;
      }
      __syncthreads();  // buffer free for reuse before next issue into it
    }
  }

  const int pw = p0 + wave * 16;  // this wave's 16-row strip

  // ---- GEMM 1: sim = (H o wp) x U^T, 16 N-tiles x 8 K-steps ---------------
  v8f zero = {0.f, 0.f, 0.f, 0.f, 0.f, 0.f, 0.f, 0.f};
  v8f acc[16];
#pragma unroll
  for (int t = 0; t < 16; ++t) acc[t] = zero;

  const float* hrow = Hn + (size_t)(pw + L) * DD;  // A: lane L owns row L
  for (int kk = 0; kk < 8; ++kk) {
    if (kk < 7) __builtin_prefetch(hrow + (kk + 1) * 32, 0, 1);
    // A fragment (contiguous K order, same as B): lane half 'hi' carries
    // K = kk*32 + hi*16 + [0..16) (contiguous)
    Frag af;
    const int k0 = kk * 32 + hi * 16;
    const int k1 = k0 + 8;
#pragma unroll
    for (int e = 0; e < 8; ++e) af.u[e]     = f2bf_bits(hrow[k0 + e] * wp_s[k0 + e]);
#pragma unroll
    for (int e = 0; e < 8; ++e) af.u[8 + e] = f2bf_bits(hrow[k1 + e] * wp_s[k1 + e]);
#pragma unroll
    for (int t = 0; t < 16; ++t) {
      // B fragment: column a = t*16+L is row a of U; lane half 'hi' carries
      // the contiguous 16 K-values kk*32 + hi*16 + [0..16)
      Frag bf;
      const unsigned short* src = &Ub[(t * 16 + L) * DD + kk * 32 + hi * 16];
      bf.q[0] = *(const Q16*)(src);
      bf.q[1] = *(const Q16*)(src + 8);
      acc[t] = wmma_bf16(af, bf, acc[t]);
    }
  }

  // ---- add hc + ua, row max (b), softmax in C-register layout -------------
  // C layout: VGPR r, lanes 0-15 -> row r, lanes 16-31 -> row r+8; col = t*16+L
  float hcv[8], uav[16], rmax[8], inv[8];
#pragma unroll
  for (int r = 0; r < 8; ++r) hcv[r] = hc_s[wave * 16 + r + hi * 8];
#pragma unroll
  for (int t = 0; t < 16; ++t) uav[t] = ua_s[t * 16 + L];
#pragma unroll
  for (int r = 0; r < 8; ++r) {
    float m = -3.402823466e38f;
#pragma unroll
    for (int t = 0; t < 16; ++t) {
      acc[t][r] += hcv[r] + uav[t];
      m = fmaxf(m, acc[t][r]);
    }
    rmax[r] = red_max16(m);                    // b[p] for this lane's row
    float s = 0.f;
#pragma unroll
    for (int t = 0; t < 16; ++t) {
      acc[t][r] = __expf(acc[t][r] - rmax[r]);
      s += acc[t][r];
    }
    inv[r] = 1.f / red_sum16(s);
  }

  // probs -> LDS (transpose C layout into A-operand row-major layout)
#pragma unroll
  for (int t = 0; t < 16; ++t)
#pragma unroll
    for (int r = 0; r < 8; ++r)
      probs[(wave * 16 + r + hi * 8) * AA + t * 16 + L] =
          f2bf_bits(acc[t][r] * inv[r]);
  __syncthreads();

  // ---- GEMM 2: U_tilda = probs x U, B from Ut -----------------------------
  v8f acc2[16];
#pragma unroll
  for (int t = 0; t < 16; ++t) acc2[t] = zero;
  const unsigned short* prow = &probs[(wave * 16 + L) * AA];
  for (int kk = 0; kk < 8; ++kk) {
    Frag af;
    af.q[0] = *(const Q16*)(prow + kk * 32 + hi * 16);
    af.q[1] = *(const Q16*)(prow + kk * 32 + hi * 16 + 8);
#pragma unroll
    for (int t = 0; t < 16; ++t) {
      Frag bf;
      const unsigned short* src = &Ut[(t * 16 + L) * AA + kk * 32 + hi * 16];
      bf.q[0] = *(const Q16*)(src);
      bf.q[1] = *(const Q16*)(src + 8);
      acc2[t] = wmma_bf16(af, bf, acc2[t]);
    }
  }

  // ---- epilogue: G[:,:,0:768] (NT stores) + deterministic b*H partials ----
#pragma unroll
  for (int t = 0; t < 16; ++t) {
    const int d = t * 16 + L;
    float htacc = 0.f;
#pragma unroll
    for (int r = 0; r < 8; ++r) {
      const int p = pw + r + hi * 8;
      const float h  = Hn[(size_t)p * DD + d];
      const float ut = acc2[t][r];
      const size_t gb = (size_t)(n * PP + p) * 1024;
      __builtin_nontemporal_store(h,      &out[gb + d]);        // slot 0: H
      __builtin_nontemporal_store(ut,     &out[gb + 256 + d]);  // slot 1
      __builtin_nontemporal_store(h * ut, &out[gb + 512 + d]);  // slot 2
      htacc += rmax[r] * h;            // b * H
    }
    htacc += swz_xor<16>(htacc);       // combine the two 8-row halves
    if (hi == 0) bh_s[wave * 256 + d] = htacc;
  }
  __syncthreads();
  for (int d = tid; d < 256; d += TPB) {
    float s = bh_s[d] + bh_s[256 + d] + bh_s[512 + d] + bh_s[768 + d];
    partial[((size_t)n * NWGP + pt) * 256 + d] = s;
  }
}

// reduce 32 per-tile partials -> H_tilda_row[n][d]
__global__ void pass2(const float* __restrict__ partial,
                      float* __restrict__ Ht) {
  const int n = blockIdx.x, d = threadIdx.x;
  float s = 0.f;
  for (int w = 0; w < NWGP; ++w) s += partial[((size_t)n * NWGP + w) * 256 + d];
  Ht[n * 256 + d] = s;
}

// G slot 3: H * broadcast(H_tilda_row)
__global__ void pass3(const float* __restrict__ H, const float* __restrict__ Ht,
                      float* __restrict__ out) {
  const size_t idx = (size_t)blockIdx.x * blockDim.x + threadIdx.x;
  const int n = (int)(idx >> 19);          // P*D = 2^19
  const int p = (int)((idx >> 8) & 2047);
  const int d = (int)(idx & 255);
  const float h = H[idx];
  __builtin_nontemporal_store(h * Ht[n * 256 + d],
                              &out[(size_t)(n * PP + p) * 1024 + 768 + d]);
}

extern "C" void kernel_launch(void* const* d_in, const int* in_sizes, int n_in,
                              void* d_out, int out_size, void* d_ws,
                              size_t ws_size, hipStream_t stream) {
  const float* H = (const float*)d_in[0];
  const float* U = (const float*)d_in[1];
  const float* W = (const float*)d_in[2];
  float* out = (float*)d_out;
  float* ws  = (float*)d_ws;
  float* partial = ws;                            // 16*32*256 f32 = 512 KB
  float* Ht      = ws + (size_t)NN * NWGP * 256;  // 16*256 f32

  pass1<<<dim3(NWGP, NN), dim3(TPB), SMEM_BYTES, stream>>>(H, U, W, out,
                                                           partial);
  pass2<<<dim3(NN), dim3(256), 0, stream>>>(partial, Ht);
  pass3<<<dim3((NN * PP * DD) / 256), dim3(256), 0, stream>>>(H, Ht, out);
}
